// WinMultiHeadSA_65257733095771
// MI455X (gfx1250) — matrix operand order repaired and checked by reading, not compiled, not run on hardware
//
#include <hip/hip_runtime.h>

#define D_DIM 128
#define HEADS 8
#define WIN 7
#define PTOK 49      // real tokens per window
#define PT 64        // padded tokens
#define NB 4
#define CW 112       // patch grid width
#define PP 12544     // patches per image (112*112)
#define WINS 256     // windows per image (16*16)

typedef __attribute__((ext_vector_type(16))) __bf16 bf16x16;
typedef __attribute__((ext_vector_type(8)))  __bf16 bf16x8;
typedef __attribute__((ext_vector_type(2)))  __bf16 bf16x2;
typedef __attribute__((ext_vector_type(8)))  float  f32x8;
typedef __attribute__((ext_vector_type(2)))  float  f32x2;

__device__ __forceinline__ unsigned short bfbits(__bf16 b) {
    union { __bf16 b; unsigned short u; } r; r.b = b; return r.u;
}

// A fragment (16x32 bf16, MxK): lane holds row M=lane&15; K chunks at
// k0..k0+7 and k0+16..k0+23 with k0 = (lane<16 ? 0 : 8).  Memory row-major [M][K].
__device__ __forceinline__ bf16x16 load_a_frag(const unsigned short* base, int ld, int lane) {
    int row = lane & 15;
    int k0  = (lane & 16) ? 8 : 0;
    const unsigned short* p = base + row * ld + k0;
    union { uint4 q[2]; bf16x16 v; } r;
    r.q[0] = *(const uint4*)(p);
    r.q[1] = *(const uint4*)(p + 16);
    return r.v;
}

// B fragment (32x16 bf16, KxN): lane holds column N=lane&15; K chunks at
// k0..k0+7 and k0+16..k0+23 with k0 = (lane<16 ? 0 : 8).  Memory laid out [N][K] (k-contiguous).
__device__ __forceinline__ bf16x16 load_b_frag(const unsigned short* base, int ld, int lane) {
    int col = lane & 15;
    int k0  = (lane & 16) ? 8 : 0;
    const unsigned short* p = base + col * ld + k0;
    union { uint4 q[2]; bf16x16 v; } r;
    r.q[0] = *(const uint4*)(p);
    r.q[1] = *(const uint4*)(p + 16);
    return r.v;
}

__device__ __forceinline__ f32x8 wmma_bf16(bf16x16 a, bf16x16 b, f32x8 c) {
    return __builtin_amdgcn_wmma_f32_16x16x32_bf16(false, a, false, b, (short)0, c, false, false);
}

// --- weight conversion: f32 -> bf16 (Wq|Wk|Wv stacked, Wo) ------------------
__global__ void convert_weights_kernel(const float* __restrict__ Wq, const float* __restrict__ Wk,
                                       const float* __restrict__ Wv, const float* __restrict__ Wo,
                                       unsigned short* __restrict__ wqkv, unsigned short* __restrict__ wo)
{
    int i = blockIdx.x * blockDim.x + threadIdx.x;
    if (i < 1024 * 128 / 2) {
        f32x2 a = *(const f32x2*)(Wq + i * 2);
        f32x2 b = *(const f32x2*)(Wk + i * 2);
        f32x2 c = *(const f32x2*)(Wv + i * 2);
        f32x2 d = *(const f32x2*)(Wo + i * 2);
        *(bf16x2*)(wqkv + i * 2)          = __builtin_convertvector(a, bf16x2);
        *(bf16x2*)(wqkv + i * 2 + 131072) = __builtin_convertvector(b, bf16x2);
        *(bf16x2*)(wqkv + i * 2 + 262144) = __builtin_convertvector(c, bf16x2);
        *(bf16x2*)(wo + i * 2)            = __builtin_convertvector(d, bf16x2);
    }
}

// --- fully fused window attention: one workgroup per (batch, window) --------
__global__ void __launch_bounds__(256) win_attn_fused(
    const float* __restrict__ x, const float* __restrict__ bq,
    const float* __restrict__ bk, const float* __restrict__ bv,
    const float* __restrict__ bo, const float* __restrict__ pos_code,
    const unsigned short* __restrict__ wqkv, const unsigned short* __restrict__ wo,
    float* __restrict__ out)
{
    extern __shared__ char smem[];
    unsigned short* sX  = (unsigned short*)(smem);           // [64][128] bf16  x window, [t][d]
    unsigned short* sQ  = (unsigned short*)(smem + 16384);   // [64][128] bf16  q, [t][d]
    unsigned short* sK  = (unsigned short*)(smem + 32768);   // [64][128] bf16  k, [t][d]
    unsigned short* sV  = (unsigned short*)(smem + 49152);   // [128][64] bf16  v, [d][t]
    float*          sS  = (float*)(smem + 65536);            // [64][64]  f32   scores, [j][i]
    unsigned short* sWb = (unsigned short*)(smem + 81920);   // [64][64]  bf16  weights, [j][i]
    unsigned short* sAO = (unsigned short*)(smem + 90112);   // [64][128] bf16  attn out, [t][d]
    // f32 staging for async x gather: overlaps sS/sWb/sAO (dead until phase 3)
    float*          sXF = (float*)(smem + 65536);            // [64][128] f32
    // total: 106496 bytes dynamic LDS

    const int tid  = threadIdx.x;
    const int lane = tid & 31;
    const int wv   = tid >> 5;                // wave 0..7
    const int nb   = blockIdx.x >> 8;
    const int wi   = blockIdx.x & 255;
    const int wy   = wi >> 4, wx = wi & 15;
    const int moff = (lane & 16) ? 8 : 0;     // C/D: VGPR r -> row moff+r
    const int ncol = lane & 15;               // C/D: lane -> column

    const f32x8 zero8 = {0.f, 0.f, 0.f, 0.f, 0.f, 0.f, 0.f, 0.f};

    // ---- phase 1: async-DMA gather of x window (HBM -> LDS), then bf16 cvt --
    const float* xin = x + (size_t)nb * D_DIM * PP;
    for (int i = tid; i < PT * D_DIM; i += 256) {
        int d = i >> 6;       // 0..127
        int t = i & 63;       // 0..63
        float* dstp = sXF + t * D_DIM + d;
        if (t < PTOK) {
            int pi = (wy * WIN + t / WIN) * CW + wx * WIN + t % WIN;
            unsigned long long ga = (unsigned long long)(xin + (size_t)d * PP + pi);
            unsigned ldsoff = (unsigned)(size_t)(void*)dstp;   // low 32b of generic = LDS offset
            asm volatile("global_load_async_to_lds_b32 %0, %1, off"
                         :: "v"(ldsoff), "v"(ga) : "memory");
        } else {
            *dstp = 0.f;      // zero-pad tokens 49..63
        }
    }
    asm volatile("s_wait_asynccnt 0" ::: "memory");
    __syncthreads();
    for (int i = tid; i < PT * D_DIM / 2; i += 256) {
        f32x2 v = *(const f32x2*)(sXF + i * 2);
        *(bf16x2*)(sX + i * 2) = __builtin_convertvector(v, bf16x2);  // v_cvt_pk_bf16_f32
    }
    __syncthreads();

    // ---- wave-invariant register state ------------------------------------
    const int ntq = wv & 3;   // this wave's token-column tile for phases 2/3
    // x B-fragments: invariant across all heads and all QKV tiles -> keep in VGPRs
    bf16x16 xb[4];
    #pragma unroll
    for (int kk = 0; kk < 4; ++kk)
        xb[kk] = load_b_frag(sX + ntq * 16 * D_DIM + kk * 32, D_DIM, lane);

    f32x8 yacc[4];
    #pragma unroll
    for (int i = 0; i < 4; ++i) yacc[i] = zero8;

    for (int h = 0; h < HEADS; ++h) {
        // prefetch next head's Wo rows for this wave (consumed in phase 6)
        if (h + 1 < HEADS)
            __builtin_prefetch(wo + (size_t)(wv * 16 + ncol) * 1024 + (h + 1) * D_DIM, 0, 1);

        // ---- phase 2: q/k/v = W_h * x_win + b  (3 GEMMs M=128 N=64 K=128) --
        // wave wv: token tile ntq, row tiles mt = (wv>>2) + {0,2,4,6}, all 3 projections
        for (int proj = 0; proj < 3; ++proj) {
            const float* bias = (proj == 0) ? bq : (proj == 1) ? bk : bv;
            for (int mi = 0; mi < 4; ++mi) {
                int mt = (wv >> 2) + mi * 2;
                const unsigned short* A = wqkv + ((size_t)proj * 1024 + h * D_DIM + mt * 16) * D_DIM;
                bf16x16 a[4];
                #pragma unroll
                for (int kk = 0; kk < 4; ++kk) a[kk] = load_a_frag(A + kk * 32, D_DIM, lane);
                f32x8 c = zero8;
                #pragma unroll
                for (int kk = 0; kk < 4; ++kk) c = wmma_bf16(a[kk], xb[kk], c);
                int mrow = mt * 16 + moff;
                f32x8 bvec = *(const f32x8*)(bias + h * D_DIM + mrow);   // 32B aligned
                bf16x8 pb = __builtin_convertvector(c + bvec, bf16x8);   // packed cvt
                if (proj < 2) {
                    // q/k stored [token][d'] -> 16B contiguous store per lane
                    unsigned short* dst = (proj == 0) ? sQ : sK;
                    *(bf16x8*)(dst + (ntq * 16 + ncol) * D_DIM + mrow) = pb;
                } else {
                    // v stored [d'][token] (transposed scatter, b16 stores)
                    union { bf16x8 v; unsigned short u[8]; } pu; pu.v = pb;
                    #pragma unroll
                    for (int r = 0; r < 8; ++r)
                        sV[(mrow + r) * PT + ntq * 16 + ncol] = pu.u[r];
                }
            }
        }
        __syncthreads();

        // ---- phase 3: S[i][j] = (k_i . q_j)/sqrt(D) + relbias, store [j][i] -
        // wave wv: j-tile ntq (Q frags shared), i-tiles mt = (wv>>2) + {0,2}
        {
            bf16x16 qb[4];
            #pragma unroll
            for (int kk = 0; kk < 4; ++kk)
                qb[kk] = load_b_frag(sQ + ntq * 16 * D_DIM + kk * 32, D_DIM, lane);
            int j  = ntq * 16 + ncol;
            int yj = j / WIN, xj = j % WIN;
            #pragma unroll
            for (int half = 0; half < 2; ++half) {
                int mt = (wv >> 2) + half * 2;
                bf16x16 a[4];
                #pragma unroll
                for (int kk = 0; kk < 4; ++kk)
                    a[kk] = load_a_frag(sK + mt * 16 * D_DIM + kk * 32, D_DIM, lane);
                f32x8 c = zero8;
                #pragma unroll
                for (int kk = 0; kk < 4; ++kk) c = wmma_bf16(a[kk], qb[kk], c);
                #pragma unroll
                for (int r = 0; r < 8; ++r) {
                    int i = mt * 16 + moff + r;
                    float vsc = -1e30f;           // mask padded keys (and pad columns)
                    if (i < PTOK && j < PTOK) {
                        int yi = i / WIN, xi = i % WIN;
                        int idx = (yi - yj + WIN - 1) + (xi - xj + WIN - 1) * (2 * WIN - 1);
                        vsc = c[r] * 0.08838834764831845f + pos_code[idx * HEADS + h];
                    }
                    c[r] = vsc;
                }
                *(f32x8*)(sS + j * PT + mt * 16 + moff) = c;   // 32B aligned
            }
        }
        __syncthreads();

        // ---- phase 4: softmax over i (contiguous per row j), -> bf16 -------
        if (tid < PT) {
            const float* row = sS + tid * PT;
            float m = row[0];
            for (int i = 1; i < PT; ++i) m = fmaxf(m, row[i]);
            float s = 0.f;
            for (int i = 0; i < PT; ++i) s += __expf(row[i] - m);
            float inv = 1.f / s;
            for (int i = 0; i < PT; ++i)
                sWb[tid * PT + i] = bfbits((__bf16)(__expf(row[i] - m) * inv));
        }
        __syncthreads();

        // ---- phase 5: ao[t][d] = sum_i w[t][i] * v[d][i]  (M=64 N=128 K=64) -
        // wave wv: d-tile wv (V frags shared), token tiles mt = 0..3
        {
            bf16x16 vb[2];
            #pragma unroll
            for (int kk = 0; kk < 2; ++kk)
                vb[kk] = load_b_frag(sV + wv * 16 * PT + kk * 32, PT, lane);
            #pragma unroll
            for (int mt = 0; mt < 4; ++mt) {
                bf16x16 a[2];
                #pragma unroll
                for (int kk = 0; kk < 2; ++kk)
                    a[kk] = load_a_frag(sWb + mt * 16 * PT + kk * 32, PT, lane);
                f32x8 c = zero8;
                c = wmma_bf16(a[0], vb[0], c);
                c = wmma_bf16(a[1], vb[1], c);
                int mrow = mt * 16 + moff;
                union { bf16x8 v; unsigned short u[8]; } pu;
                pu.v = __builtin_convertvector(c, bf16x8);
                #pragma unroll
                for (int r = 0; r < 8; ++r)
                    sAO[(mrow + r) * D_DIM + wv * 16 + ncol] = pu.u[r];
            }
        }
        __syncthreads();

        // ---- phase 6: y += Wo[:, h*128:(h+1)*128] * ao^T (persistent accum) -
        // wave wv owns output rows [wv*16, wv*16+16); Wo A-frags hoisted (nt-invariant)
        {
            bf16x16 aw[4];
            #pragma unroll
            for (int kk = 0; kk < 4; ++kk)
                aw[kk] = load_a_frag(wo + (size_t)(wv * 16) * 1024 + h * D_DIM + kk * 32, 1024, lane);
            #pragma unroll
            for (int nt = 0; nt < 4; ++nt) {
                bf16x16 b[4];
                #pragma unroll
                for (int kk = 0; kk < 4; ++kk)
                    b[kk] = load_b_frag(sAO + nt * 16 * D_DIM + kk * 32, D_DIM, lane);
                #pragma unroll
                for (int kk = 0; kk < 4; ++kk)
                    yacc[nt] = wmma_bf16(aw[kk], b[kk], yacc[nt]);
            }
        }
        __syncthreads();   // protect sQ/sK/sV/sAO for next head
    }

    // ---- phase 7: y + bo scattered back to [N][128][P] ---------------------
    float* yout = out + (size_t)nb * D_DIM * PP;
    f32x8 bovec = *(const f32x8*)(bo + wv * 16 + moff);   // 32B aligned
    for (int nt = 0; nt < 4; ++nt) {
        int t = nt * 16 + ncol;
        if (t < PTOK) {
            int pi = (wy * WIN + t / WIN) * CW + wx * WIN + t % WIN;
            f32x8 yv = yacc[nt] + bovec;
            #pragma unroll
            for (int r = 0; r < 8; ++r) {
                int o = wv * 16 + moff + r;
                yout[(size_t)o * PP + pi] = yv[r];
            }
        }
    }
}

extern "C" void kernel_launch(void* const* d_in, const int* in_sizes, int n_in,
                              void* d_out, int out_size, void* d_ws, size_t ws_size,
                              hipStream_t stream) {
    (void)in_sizes; (void)n_in; (void)out_size; (void)ws_size;
    const float* x   = (const float*)d_in[0];
    const float* Wq  = (const float*)d_in[1];
    const float* bq  = (const float*)d_in[2];
    const float* Wk  = (const float*)d_in[3];
    const float* bk  = (const float*)d_in[4];
    const float* Wv  = (const float*)d_in[5];
    const float* bv  = (const float*)d_in[6];
    const float* Wo  = (const float*)d_in[7];
    const float* bo  = (const float*)d_in[8];
    const float* pos = (const float*)d_in[9];
    // d_in[10], d_in[11] = h, w scalars: compile-time constants here

    // ws: bf16 Wqkv [3][1024][128] + bf16 Wo [128][1024]  (~1.05 MB)
    unsigned short* wqkv = (unsigned short*)d_ws;
    unsigned short* wo   = wqkv + 3 * 1024 * 128;
    float* out = (float*)d_out;

    convert_weights_kernel<<<256, 256, 0, stream>>>(Wq, Wk, Wv, Wo, wqkv, wo);

    const int lds = 106496;   // > 64KB default: raise dynamic-LDS cap (WGP has 320KB)
    (void)hipFuncSetAttribute(reinterpret_cast<const void*>(win_attn_fused),
                              hipFuncAttributeMaxDynamicSharedMemorySize, lds);
    win_attn_fused<<<NB * WINS, 256, lds, stream>>>(x, bq, bk, bv, bo, pos, wqkv, wo, out);
}
